// Attention_57767310131389
// MI455X (gfx1250) — compile-verified, matrix operand order checked by reading, not run on hardware
//
#include <hip/hip_runtime.h>
#include <stdint.h>

// ---------------------------------------------------------------------------
// Gated multi-head attention for MI455X (gfx1250), bf16 WMMA pipeline.
// B=8, N=1568, C=768, H=12, HD=64.  All matmuls via v_wmma_f32_16x16x32_bf16.
// Everything is L2-resident (<192MB); kernels use coalesced b128 global
// staging + contiguous b64/b128 LDS fragment loads, 32-bit addressing, and
// batched cross-lane reductions (8 independent bpermutes per wait).
// ---------------------------------------------------------------------------

typedef __attribute__((ext_vector_type(16))) __bf16 v16bf;
typedef __attribute__((ext_vector_type(8)))  float  v8f;

#define WMMA_BF16(a, b, c)                                                    \
  __builtin_amdgcn_wmma_f32_16x16x32_bf16(false, (a), false, (b), (short)0,   \
                                          (c), false, false)

static constexpr int B_  = 8;
static constexpr int N_  = 1568;   // 98*16 = 49*32
static constexpr int C_  = 768;
static constexpr int H_  = 12;
static constexpr int HD_ = 64;
static constexpr int C3_ = 3 * C_; // 2304
static constexpr float SCALE_ = 0.125f; // 64^-0.5

// Fragment register image: v16bf == 8 VGPRs; VGPR pair (2g,2g+1) holds the 4
// contiguous K values starting at kbase4(hf,g) (16-bit 16x32 layout, ISA
// 7.12.2: lanes 0-15 khalf=0, lanes 16-31 khalf=8).
union Frag {
  v16bf v;
  uint2 u2[4];
};

__device__ __forceinline__ int kbase4(int hf, int g) {
  return (hf << 3) + ((g & 1) << 2) + ((g >> 1) << 4);
}

// ---------------------------------------------------------------------------
// Tiled GEMM: out[m,n] = sum_k A[m,k] * W_f32[n,k] (+bias[n])
// Block = 128 threads (4 waves) computing a 64x64 tile; A/W panels (64x32)
// staged through LDS as bf16 with coalesced 16B loads.
// grid = (Nout/64, M/64).
// ---------------------------------------------------------------------------
template <typename AT, typename OT, bool HAS_BIAS>
__global__ __launch_bounds__(128) void gemm_wmma_kernel(
    const AT* __restrict__ A, const float* __restrict__ W,
    const float* __restrict__ bias, OT* __restrict__ out,
    int M, int K, int Nout) {
  __shared__ __bf16 aBuf[64 * 32];
  __shared__ __bf16 wBuf[64 * 32];

  const int nt   = blockIdx.x;   // 64-wide N tile
  const int mt   = blockIdx.y;   // 64-tall M tile
  const int tid  = threadIdx.x;  // 0..127
  const int wv   = tid >> 5;     // wave id 0..3 -> M subtile
  const int lane = tid & 31;
  const int hf   = lane >> 4;
  const int mr   = lane & 15;

  v8f acc[4] = {};

  for (int kk = 0; kk < K; kk += 32) {
    // ---- stage A panel (64 rows x 32 k) ----
    if constexpr (sizeof(AT) == 4) {
#pragma unroll
      for (int p = 0; p < 4; ++p) {
        const int row = p * 16 + (tid >> 3);
        const int kq  = (tid & 7) * 4;
        const float4 v = *(const float4*)(A + (mt * 64 + row) * K + kk + kq);
        __bf16* dst = &aBuf[row * 32 + kq];
        dst[0] = (__bf16)v.x; dst[1] = (__bf16)v.y;
        dst[2] = (__bf16)v.z; dst[3] = (__bf16)v.w;
      }
    } else {
#pragma unroll
      for (int p = 0; p < 2; ++p) {
        const int row  = p * 32 + (tid >> 2);
        const int coff = (tid & 3) * 8;  // 8 bf16 = 16B
        const uint4 v = *(const uint4*)(A + (mt * 64 + row) * K + kk + coff);
        *(uint4*)(&aBuf[row * 32 + coff]) = v;
      }
    }
    // ---- stage W panel (64 rows x 32 k), always f32 source ----
#pragma unroll
    for (int p = 0; p < 4; ++p) {
      const int row = p * 16 + (tid >> 3);
      const int kq  = (tid & 7) * 4;
      const float4 v = *(const float4*)(W + (nt * 64 + row) * K + kk + kq);
      __bf16* dst = &wBuf[row * 32 + kq];
      dst[0] = (__bf16)v.x; dst[1] = (__bf16)v.y;
      dst[2] = (__bf16)v.z; dst[3] = (__bf16)v.w;
    }
    __syncthreads();

    // ---- fragments from LDS (b64/b128 loads); A reused over 4 B frags ----
    Frag af;
#pragma unroll
    for (int g = 0; g < 4; ++g)
      af.u2[g] = *(const uint2*)(&aBuf[(wv * 16 + mr) * 32 + kbase4(hf, g)]);
#pragma unroll
    for (int c = 0; c < 4; ++c) {
      Frag bf;
#pragma unroll
      for (int g = 0; g < 4; ++g)
        bf.u2[g] = *(const uint2*)(&wBuf[(c * 16 + mr) * 32 + kbase4(hf, g)]);
      acc[c] = WMMA_BF16(af.v, bf.v, acc[c]);
    }
    __syncthreads();
  }

  // ---- write 4 tiles: D col = lane&15, row = j + 8*(lane>>4) ----
#pragma unroll
  for (int c = 0; c < 4; ++c) {
    const int col = nt * 64 + c * 16 + mr;
    float bv = 0.0f;
    if constexpr (HAS_BIAS) bv = bias[col];
#pragma unroll
    for (int j = 0; j < 8; ++j) {
      const int row = mt * 64 + wv * 16 + j + 8 * hf;
      if constexpr (HAS_BIAS)
        out[row * Nout + col] = (OT)(acc[c][j] + bv);
      else
        out[row * Nout + col] = (OT)acc[c][j];
    }
  }
}

// ---------------------------------------------------------------------------
// Gate kernel: sigmoid gate over concat(q,k,v) per (b,h,n); emit gated Q and
// gated K in natural [B,H,N,HD] order, V transposed to [B,H,HD,N].
// Block = 256 threads = 4 tokens x 64 head-dim elements.
// ---------------------------------------------------------------------------
__global__ __launch_bounds__(256) void gate_kernel(
    const __bf16* __restrict__ qkv,  // [B,N,3C]
    const float* __restrict__ pw,    // [HD, 3*HD]
    const float* __restrict__ pb,    // [HD]
    __bf16* __restrict__ qg,         // [B,H,N,HD]
    __bf16* __restrict__ kg,         // [B,H,N,HD]
    __bf16* __restrict__ vT) {       // [B,H,HD,N]
  const int sub = threadIdx.x >> 6;          // token slot 0..3
  const int e   = threadIdx.x & 63;          // head-dim element
  const int idx = blockIdx.x * 4 + sub;      // bh * N + n
  const int n   = idx % N_;
  const int bh  = idx / N_;
  const int b   = bh / H_;
  const int h   = bh % H_;

  __shared__ __bf16 cat[4][3 * HD_];

  const int rowbase = (b * N_ + n) * C3_ + h * HD_;
  cat[sub][e]           = qkv[rowbase + e];
  cat[sub][HD_ + e]     = qkv[rowbase + C_ + e];
  cat[sub][2 * HD_ + e] = qkv[rowbase + 2 * C_ + e];
  __syncthreads();

  float s = pb[e];
  const float* __restrict__ wrow = pw + e * (3 * HD_);
#pragma unroll 8
  for (int d = 0; d < 3 * HD_; ++d) s += (float)cat[sub][d] * wrow[d];
  const float g = 1.0f / (1.0f + __expf(-s));

  qg[(bh * N_ + n) * HD_ + e] = (__bf16)(g * (float)cat[sub][e]);
  kg[(bh * N_ + n) * HD_ + e] = (__bf16)(g * (float)cat[sub][HD_ + e]);
  vT[(bh * HD_ + e) * N_ + n] = cat[sub][2 * HD_ + e];
}

// ---------------------------------------------------------------------------
// Flash attention: block = 64 threads (2 waves), each wave owns 16 queries;
// waves share LDS-staged K (32 keys x 64 hd) and V^T (64 hd x 32 keys) tiles.
// grid = (N/32, B*H).
// Layouts chosen so all WMMA fragment pairs are contiguous -> b64+ LDS loads:
//   S  B-frag pairs run along hd  -> kBuf [key][hd]
//   PV B-frag pairs run along key -> vBufT [hd][key]
//   P  A-frag pairs run along key -> pBuf [row][key]
// Staging offsets linearize: LDS elem off == chunk*8; global off == base+ch*8.
// ---------------------------------------------------------------------------
__global__ __launch_bounds__(64) void attn_kernel(
    const __bf16* __restrict__ qg,  // [B,H,N,HD]
    const __bf16* __restrict__ kg,  // [B,H,N,HD]
    const __bf16* __restrict__ vT,  // [B,H,HD,N]
    __bf16* __restrict__ AO) {      // [B,N,C]
  __shared__ __bf16 kBuf[32 * 64];      // [key][hd]
  __shared__ __bf16 vBufT[64 * 32];     // [hd][key]
  __shared__ __bf16 pBuf[2][16 * 32];   // per-wave P tile [row][key]

  const int qt2  = blockIdx.x;          // 32-query super-tile
  const int bh   = blockIdx.y;
  const int b    = bh / H_;
  const int h    = bh % H_;
  const int tid  = threadIdx.x;         // 0..63
  const int wv   = tid >> 5;            // wave -> 16-query subtile
  const int lane = tid & 31;
  const int hf   = lane >> 4;
  const int mr   = lane & 15;

  const __bf16* __restrict__ kbase = kg + bh * (N_ * HD_);
  const __bf16* __restrict__ vbase = vT + bh * (HD_ * N_);

  // Q A-fragments for the two 32-wide HD chunks (b64/b128 global loads).
  const int qoff = (bh * N_ + qt2 * 32 + wv * 16 + mr) * HD_;
  Frag qa[2];
#pragma unroll
  for (int c = 0; c < 2; ++c)
#pragma unroll
    for (int g = 0; g < 4; ++g)
      qa[c].u2[g] = *(const uint2*)(qg + qoff + c * 32 + kbase4(hf, g));

  float mrow[8], lrow[8];
#pragma unroll
  for (int j = 0; j < 8; ++j) { mrow[j] = -1e30f; lrow[j] = 0.0f; }
  v8f o[4] = {};

  // loop-invariant staging offsets (chunk ch = it*64 + tid; elem off = ch*8)
  const int kgo = tid * 8;                        // + kb*64 + it*512 (global)
  const int vgo = (tid >> 2) * N_ + (tid & 3) * 8;  // + kb + it*16*N_ (global)
  __bf16* const pw0 = &pBuf[wv][(8 * hf) * 32 + mr];  // + j*32 (+16 for p1)

  for (int kb = 0; kb < N_; kb += 32) {
    // ---- cooperative staging (coalesced 16B chunks) ----
#pragma unroll
    for (int it = 0; it < 4; ++it)      // K tile: 32 keys x 64 hd
      *(uint4*)(&kBuf[kgo + it * 512]) =
          *(const uint4*)(kbase + kb * 64 + kgo + it * 512);
#pragma unroll
    for (int it = 0; it < 4; ++it)      // V^T tile: 64 hd x 32 keys
      *(uint4*)(&vBufT[tid * 8 + it * 512]) =
          *(const uint4*)(vbase + vgo + it * (16 * N_) + kb);
    __syncthreads();

    // ---- S = Q K^T for 32 keys (two 16-key subtiles) ----
    v8f s0 = {}, s1 = {};
#pragma unroll
    for (int c = 0; c < 2; ++c) {
      Frag b0, b1;
#pragma unroll
      for (int g = 0; g < 4; ++g) {
        const int kd = c * 32 + kbase4(hf, g);
        b0.u2[g] = *(const uint2*)(&kBuf[mr * 64 + kd]);
        b1.u2[g] = *(const uint2*)(&kBuf[(16 + mr) * 64 + kd]);
      }
      s0 = WMMA_BF16(qa[c].v, b0.v, s0);
      s1 = WMMA_BF16(qa[c].v, b1.v, s1);
    }

    // ---- online softmax; batched cross-lane reductions (width 16) ----
    float red[8], alpha[8];
#pragma unroll
    for (int j = 0; j < 8; ++j) {
      s0[j] *= SCALE_;
      s1[j] *= SCALE_;
      red[j] = fmaxf(s0[j], s1[j]);
    }
#pragma unroll
    for (int off = 1; off < 16; off <<= 1) {
      float t[8];
#pragma unroll
      for (int j = 0; j < 8; ++j) t[j] = __shfl_xor(red[j], off, 16);
#pragma unroll
      for (int j = 0; j < 8; ++j) red[j] = fmaxf(red[j], t[j]);
    }
#pragma unroll
    for (int j = 0; j < 8; ++j) {
      const float mnew = fmaxf(mrow[j], red[j]);
      alpha[j] = __expf(mrow[j] - mnew);
      s0[j] = __expf(s0[j] - mnew);
      s1[j] = __expf(s1[j] - mnew);
      mrow[j] = mnew;
      red[j] = s0[j] + s1[j];
    }
#pragma unroll
    for (int off = 1; off < 16; off <<= 1) {
      float t[8];
#pragma unroll
      for (int j = 0; j < 8; ++j) t[j] = __shfl_xor(red[j], off, 16);
#pragma unroll
      for (int j = 0; j < 8; ++j) red[j] += t[j];
    }
#pragma unroll
    for (int j = 0; j < 8; ++j) {
      lrow[j] = lrow[j] * alpha[j] + red[j];
      o[0][j] *= alpha[j]; o[1][j] *= alpha[j];
      o[2][j] *= alpha[j]; o[3][j] *= alpha[j];
      pw0[j * 32]      = (__bf16)s0[j];
      pw0[j * 32 + 16] = (__bf16)s1[j];
    }

    // ---- A-fragment of P from per-wave LDS (same-wave DS is in-order) ----
    Frag pa;
#pragma unroll
    for (int g = 0; g < 4; ++g)
      pa.u2[g] = *(const uint2*)(&pBuf[wv][mr * 32 + kbase4(hf, g)]);

    // ---- O += P @ V over 4 hd chunks ----
#pragma unroll
    for (int c = 0; c < 4; ++c) {
      Frag vb;
#pragma unroll
      for (int g = 0; g < 4; ++g)
        vb.u2[g] = *(const uint2*)(&vBufT[(c * 16 + mr) * 32 + kbase4(hf, g)]);
      o[c] = WMMA_BF16(pa.v, vb.v, o[c]);
    }
    __syncthreads();  // all waves done with kBuf/vBufT before restaging
  }

  // ---- normalize and scatter into [B,N,C] (head h occupies cols h*64..) ----
#pragma unroll
  for (int j = 0; j < 8; ++j) {
    const float inv = 1.0f / lrow[j];
    const int row   = qt2 * 32 + wv * 16 + j + 8 * hf;
    const int ob    = (b * N_ + row) * C_ + h * HD_;
#pragma unroll
    for (int c = 0; c < 4; ++c)
      AO[ob + c * 16 + mr] = (__bf16)(o[c][j] * inv);
  }
}

// ---------------------------------------------------------------------------
// Host-side launch
// ---------------------------------------------------------------------------
static inline size_t align256(size_t x) { return (x + 255) & ~(size_t)255; }

extern "C" void kernel_launch(void* const* d_in, const int* in_sizes, int n_in,
                              void* d_out, int out_size, void* d_ws,
                              size_t ws_size, hipStream_t stream) {
  const float* x       = (const float*)d_in[0];  // [B,N,C]
  const float* qkv_w   = (const float*)d_in[1];  // [3C,C]
  const float* pgate_w = (const float*)d_in[2];  // [HD,3HD]
  const float* pgate_b = (const float*)d_in[3];  // [HD]
  const float* proj_w  = (const float*)d_in[4];  // [C,C]
  const float* proj_b  = (const float*)d_in[5];  // [C]
  float* out = (float*)d_out;                    // [B,N,C] f32

  const size_t n_qkv  = (size_t)B_ * N_ * C3_;        // 28.9M elems
  const size_t n_head = (size_t)B_ * H_ * N_ * HD_;   // 9.63M elems
  const size_t n_ao   = (size_t)B_ * N_ * C_;         // 9.63M elems

  char* p = (char*)d_ws;
  __bf16* qkv = (__bf16*)p; p += align256(n_qkv * sizeof(__bf16));
  __bf16* qg  = (__bf16*)p; p += align256(n_head * sizeof(__bf16));
  __bf16* kg  = (__bf16*)p; p += align256(n_head * sizeof(__bf16));
  __bf16* vT  = (__bf16*)p; p += align256(n_head * sizeof(__bf16));
  __bf16* ao  = (__bf16*)p; p += align256(n_ao * sizeof(__bf16));

  const int M = B_ * N_;  // 12544

  // qkv = x @ qkv_w.T   (f32 in, bf16 out), M=12544 K=768 N=2304
  gemm_wmma_kernel<float, __bf16, false>
      <<<dim3(C3_ / 64, M / 64), 128, 0, stream>>>(
          x, qkv_w, nullptr, qkv, M, C_, C3_);

  // per-head gate + layout shuffle
  gate_kernel<<<(B_ * H_ * N_) / 4, 256, 0, stream>>>(
      qkv, pgate_w, pgate_b, qg, kg, vT);

  // flash attention
  attn_kernel<<<dim3(N_ / 32, B_ * H_), 64, 0, stream>>>(qg, kg, vT, ao);

  // out = ao @ proj_w.T + proj_b  (bf16 in, f32 out), M=12544 K=768 N=768
  gemm_wmma_kernel<__bf16, float, true>
      <<<dim3(C_ / 64, M / 64), 128, 0, stream>>>(
          ao, proj_w, proj_b, out, M, C_, C_);
}
